// FeedForward_7335804141841
// MI455X (gfx1250) — 8-bit matrix operand order repaired and checked by reading, compiled, not run on hardware
//
#include <hip/hip_runtime.h>
#include <hip/hip_bf16.h>
#include <stdint.h>


typedef int v8i __attribute__((ext_vector_type(8)));

#define CDIV(a, b) (((a) + (b) - 1) / (b))

constexpr int KD = 2048;
constexpr int NH = 8192;


__device__ __forceinline__ float block_reduce_sum(float v, float* red) {
  const int tid = threadIdx.x;
  red[tid] = v; __syncthreads();
  for (int o = 128; o > 0; o >>= 1) {
    if (tid < o) red[tid] += red[tid + o];
    __syncthreads();
  }
  float r = red[0]; __syncthreads();
  return r;
}

__device__ __forceinline__ float block_reduce_max(float v, float* red) {
  const int tid = threadIdx.x;
  red[tid] = v; __syncthreads();
  for (int o = 128; o > 0; o >>= 1) {
    if (tid < o) red[tid] = fmaxf(red[tid], red[tid + o]);
    __syncthreads();
  }
  float r = red[0]; __syncthreads();
  return r;
}

__device__ __forceinline__ int q8(float v, float s) {
  return (int)fminf(127.0f, fmaxf(-128.0f, rintf(v * s)));
}

__device__ __forceinline__ v8i load_a_frag(const int8_t* ab) {
  int2 a01 = *(const int2*)(ab + 0);
  int2 a23 = *(const int2*)(ab + 16);
  int2 a45 = *(const int2*)(ab + 32);
  int2 a67 = *(const int2*)(ab + 48);
  v8i av;
  av[0] = a01.x; av[1] = a01.y; av[2] = a23.x; av[3] = a23.y;
  av[4] = a45.x; av[5] = a45.y; av[6] = a67.x; av[7] = a67.y;
  return av;
}

__device__ __forceinline__ v8i load_b_frag(const int8_t* bb) {
  int2 b01 = *(const int2*)(bb + 0);
  int2 b23 = *(const int2*)(bb + 16);
  int2 b45 = *(const int2*)(bb + 32);
  int2 b67 = *(const int2*)(bb + 48);
  v8i bv;
  bv[0] = b01.x; bv[1] = b01.y; bv[2] = b23.x; bv[3] = b23.y;
  bv[4] = b45.x; bv[5] = b45.y; bv[6] = b67.x; bv[7] = b67.y;
  return bv;
}


__global__ void absmean_partial_kernel(const float* __restrict__ w,
                                       float* __restrict__ partial,
                                       long long n) {
  __shared__ float red[256];
  float s = 0.0f;
  const long long stride = (long long)gridDim.x * blockDim.x;
  for (long long i = (long long)blockIdx.x * blockDim.x + threadIdx.x; i < n; i += stride)
    s += fabsf(w[i]);
  s = block_reduce_sum(s, red);
  if (threadIdx.x == 0) partial[blockIdx.x] = s;
}

__global__ void absmean_final_kernel(const float* __restrict__ partial,
                                     float* __restrict__ wscale,
                                     long long n) {
  __shared__ float red[256];
  float s = 0.0f;
  for (int i = threadIdx.x; i < 1024; i += 256) s += partial[i];
  s = block_reduce_sum(s, red);
  if (threadIdx.x == 0) wscale[0] = fmaxf(s / (float)n, 1e-5f);
}

__global__ void weight_quant_kernel(const float* __restrict__ w,
                                    const float* __restrict__ wscale,
                                    int8_t* __restrict__ q, long long n) {
  const float inv = 1.0f / wscale[0];
  const long long stride = (long long)gridDim.x * blockDim.x;
  for (long long i = (long long)blockIdx.x * blockDim.x + threadIdx.x; i < n; i += stride) {
    float r = rintf(w[i] * inv);
    r = fminf(1.0f, fmaxf(-1.0f, r));
    q[i] = (int8_t)(int)r;
  }
}


__global__ void ln_quant_kernel(const float* __restrict__ x,
                                const float* __restrict__ gamma,
                                const float* __restrict__ beta,
                                int8_t* __restrict__ xq,
                                float* __restrict__ inv_sx, int D) {
  __shared__ float red[256];
  const int token = blockIdx.x;
  const int tid = threadIdx.x;
  const int c0 = tid * 8;
  const float* xr = x + (size_t)token * D;

  float v[8];
  *(float4*)&v[0] = *(const float4*)(xr + c0);
  *(float4*)&v[4] = *(const float4*)(xr + c0 + 4);

  float s = 0.0f, ss = 0.0f;
#pragma unroll
  for (int j = 0; j < 8; ++j) { s += v[j]; ss += v[j] * v[j]; }
  s  = block_reduce_sum(s, red);
  ss = block_reduce_sum(ss, red);
  const float mu   = s / (float)D;
  const float var  = ss / (float)D - mu * mu;
  const float rstd = rsqrtf(var + 1e-5f);

  float amax = 0.0f;
#pragma unroll
  for (int j = 0; j < 8; ++j) {
    v[j] = (v[j] - mu) * rstd * gamma[c0 + j] + beta[c0 + j];
    amax = fmaxf(amax, fabsf(v[j]));
  }
  amax = block_reduce_max(amax, red);
  const float cmax = fmaxf(amax, 1e-5f);
  const float sq = 127.0f / cmax;

  int lo = 0, hi = 0;
#pragma unroll
  for (int j = 0; j < 4; ++j) lo |= ((int)(unsigned char)q8(v[j], sq)) << (8 * j);
#pragma unroll
  for (int j = 0; j < 4; ++j) hi |= ((int)(unsigned char)q8(v[4 + j], sq)) << (8 * j);
  int2 pk; pk.x = lo; pk.y = hi;
  *(int2*)(xq + (size_t)token * D + c0) = pk;

  if (tid == 0) inv_sx[token] = cmax / 127.0f;
}

__global__ __launch_bounds__(512) void ffn_fused_kernel(
    const int8_t* __restrict__ Axq, const int8_t* __restrict__ w1q,
    const int8_t* __restrict__ w2q, const float* __restrict__ inv_sx,
    const float* __restrict__ wsc1, const float* __restrict__ wsc2,
    const float* __restrict__ b1, const float* __restrict__ b2,
    float* __restrict__ out) {
  constexpr int A_BYTES = 16 * KD;
  constexpr int H_BYTES = 16 * NH * 2;
  __shared__ char  smem[A_BYTES + H_BYTES];
  __shared__ float smax[16][33];
  __shared__ float sinv[16];
  __shared__ float sdq[16];

  int8_t*   a_lds = (int8_t*)smem;
  _Float16* h_lds = (_Float16*)(smem + A_BYTES);
  int8_t*   hq_lds = (int8_t*)smem;
  int*      hq32   = (int*)smem;

  const int tid  = threadIdx.x;
  const int lane = tid & 31;
  const int wave = tid >> 5;
  const int half = lane >> 4;
  const int l16  = lane & 15;
  const int m_base = blockIdx.x * 16;

  for (int i = tid; i < 16 * KD / 16; i += 512) {
    const int row = i >> 7;
    const int col = (i & 127) * 16;
    *(int4*)&a_lds[row * KD + col] =
        *(const int4*)(Axq + (size_t)(m_base + row) * KD + col);
  }
  __syncthreads();

  const float sw1 = wsc1[0];
  float isx[8];
#pragma unroll
  for (int r = 0; r < 8; ++r) isx[r] = inv_sx[m_base + half * 8 + r];

  for (int nt = 0; nt < NH / (16 * 16); ++nt) {
    const int ncol = wave * (NH / 16) + nt * 16 + l16;
    const int8_t* Brow = w1q + (size_t)ncol * KD;

    v8i acc = {0, 0, 0, 0, 0, 0, 0, 0};
#pragma unroll 4
    for (int kk = 0; kk < KD; kk += 64) {
      v8i av = load_a_frag(&a_lds[l16 * KD + kk + half * 8]);
      v8i bv = load_b_frag(Brow + kk + half * 8);
      acc = __builtin_amdgcn_wmma_i32_16x16x64_iu8(true, av, true, bv, acc,
                                                   false, false);
    }

    const float bn = b1[ncol];
#pragma unroll
    for (int r = 0; r < 8; ++r) {
      const int ml = r + half * 8;
      float v = (float)acc[r] * isx[r] * sw1 + bn;
      float g = 0.5f * v * (1.0f + erff(v * 0.70710678118654752f));
      h_lds[ml * NH + ncol] = (_Float16)g;
    }
  }
  __syncthreads();

  {
    const int trow = tid >> 5;
    const int tcol = tid & 31;
    float am = 0.0f;
    for (int c = tcol; c < NH; c += 32)
      am = fmaxf(am, fabsf((float)h_lds[trow * NH + c]));
    smax[trow][tcol] = am;
  }
  __syncthreads();
  if ((tid & 31) == 0) {
    const int trow = tid >> 5;
    float mm = 0.0f;
#pragma unroll
    for (int j = 0; j < 32; ++j) mm = fmaxf(mm, smax[trow][j]);
    const float cm = fmaxf(mm, 1e-5f);
    sinv[trow] = 127.0f / cm;
    sdq[trow]  = cm / 127.0f;
  }
  __syncthreads();

  int stash[64];
#pragma unroll
  for (int s = 0; s < 64; ++s) {
    const int i   = s * 512 + tid;
    const int e   = i * 4;
    const int tok = e >> 13;
    const int col = e & (NH - 1);
    const float sq = sinv[tok];
    const _Float16* hp = &h_lds[tok * NH + col];
    int p = 0;
#pragma unroll
    for (int j = 0; j < 4; ++j)
      p |= ((int)(unsigned char)q8((float)hp[j], sq)) << (8 * j);
    stash[s] = p;
  }
  __syncthreads();

#pragma unroll
  for (int s = 0; s < 64; ++s) hq32[s * 512 + tid] = stash[s];
  __syncthreads();

  const float sw2 = wsc2[0];
  float dq[8];
#pragma unroll
  for (int r = 0; r < 8; ++r) dq[r] = sdq[half * 8 + r];

  for (int nt = 0; nt < KD / (16 * 16); ++nt) {
    const int ncol = wave * (KD / 16) + nt * 16 + l16;
    const int8_t* Brow = w2q + (size_t)ncol * NH;

    v8i acc = {0, 0, 0, 0, 0, 0, 0, 0};
#pragma unroll 4
    for (int kk = 0; kk < NH; kk += 64) {
      v8i av = load_a_frag(&hq_lds[l16 * NH + kk + half * 8]);
      v8i bv = load_b_frag(Brow + kk + half * 8);
      acc = __builtin_amdgcn_wmma_i32_16x16x64_iu8(true, av, true, bv, acc,
                                                   false, false);
    }

    const float bn = b2[ncol];
#pragma unroll
    for (int r = 0; r < 8; ++r) {
      const int m = m_base + r + half * 8;
      out[(size_t)m * KD + ncol] = (float)acc[r] * dq[r] * sw2 + bn;
    }
  }
}


extern "C" void kernel_launch(void* const* d_in, const int* in_sizes, int n_in,
                              void* d_out, int out_size, void* d_ws,
                              size_t ws_size, hipStream_t stream) {
  const float* x     = (const float*)d_in[0];
  const float* gamma = (const float*)d_in[1];
  const float* beta  = (const float*)d_in[2];
  const float* w1    = (const float*)d_in[3];
  const float* b1    = (const float*)d_in[4];
  const float* w2    = (const float*)d_in[5];
  const float* b2    = (const float*)d_in[6];

  const int D = in_sizes[1];
  const int N = in_sizes[0] / D;
  const long long n1 = (long long)in_sizes[3];
  const long long n2 = (long long)in_sizes[5];

  char* ws = (char*)d_ws;
  size_t off = 0;
  auto alloc = [&](size_t bytes) -> char* {
    char* p = ws + off;
    off += (bytes + 255) & ~(size_t)255;
    return p;
  };
  int8_t* w1q    = (int8_t*)alloc((size_t)n1);
  int8_t* w2q    = (int8_t*)alloc((size_t)n2);
  int8_t* xq     = (int8_t*)alloc((size_t)N * D);
  float*  inv_sx = (float*)alloc((size_t)N * sizeof(float));
  float*  part1  = (float*)alloc(1024 * sizeof(float));
  float*  part2  = (float*)alloc(1024 * sizeof(float));
  float*  wsc1   = (float*)alloc(sizeof(float));
  float*  wsc2   = (float*)alloc(sizeof(float));

  absmean_partial_kernel<<<1024, 256, 0, stream>>>(w1, part1, n1);
  absmean_final_kernel<<<1, 256, 0, stream>>>(part1, wsc1, n1);
  weight_quant_kernel<<<2048, 256, 0, stream>>>(w1, wsc1, w1q, n1);

  absmean_partial_kernel<<<1024, 256, 0, stream>>>(w2, part2, n2);
  absmean_final_kernel<<<1, 256, 0, stream>>>(part2, wsc2, n2);
  weight_quant_kernel<<<2048, 256, 0, stream>>>(w2, wsc2, w2q, n2);

  ln_quant_kernel<<<N, 256, 0, stream>>>(x, gamma, beta, xq, inv_sx, D);

  ffn_fused_kernel<<<N / 16, 512, 0, stream>>>(xq, w1q, w2q, inv_sx, wsc1,
                                               wsc2, b1, b2, (float*)d_out);
}
